// FusionAttentionModel_17068200034932
// MI455X (gfx1250) — matrix operand order repaired and checked by reading, not compiled, not run on hardware
//
#include <hip/hip_runtime.h>
#include <hip/hip_bf16.h>
#include <math.h>

typedef __attribute__((ext_vector_type(16))) __bf16 v16bf;
typedef __attribute__((ext_vector_type(8)))  float  v8f;

#define BATCH 8192

union Frag  { v16bf v; float4 f[2]; };
union Pack4 { __bf16 h[4]; unsigned long long u; };

// ---------------------------------------------------------------------------
// GEMM: C[row,col] = sum_k A[row,k] * W[col,k] + bias[col]
// W: [N][K] bf16 (pre-converted).  AMODE=0: A fp32 (reg-pipelined cvt),
// AMODE=1: A bf16 (async global->LDS).  C: fp32 or bf16 (one ptr non-null).
// 128x128 tile, K-step 32, double-buffered LDS, 8 waves.
// ---------------------------------------------------------------------------
template <int AMODE>
__global__ __launch_bounds__(256) void gemm_wmma(
    const void* __restrict__ Avoid, int lda,
    const __bf16* __restrict__ Wb,
    const float* __restrict__ bias,
    float* __restrict__ Cf, __bf16* __restrict__ Cb, int ldc,
    int K, int relu)
{
  __shared__ __bf16 As[2][128][40];
  __shared__ __bf16 Bs[2][128][40];

  const float*  A32 = (const float*)Avoid;
  const __bf16* A16 = (const __bf16*)Avoid;

  const int tid  = threadIdx.x;
  const int lane = tid & 31;
  const int wid  = tid >> 5;
  const int wm   = wid & 3;       // rows wm*32..+31
  const int wn   = wid >> 2;      // cols wn*64..+63
  const int half = lane >> 4;
  const int l16  = lane & 15;

  const int rowBase = blockIdx.y * 128;
  const int colBase = blockIdx.x * 128;

  const int lrow = tid >> 3;       // 0..31 (fp32 A staging)
  const int lcol = (tid & 7) * 4;  // 0..28

  v8f zero = {0.f,0.f,0.f,0.f,0.f,0.f,0.f,0.f};
  v8f acc[2][4];
  #pragma unroll
  for (int i = 0; i < 2; ++i)
    #pragma unroll
    for (int j = 0; j < 4; ++j) acc[i][j] = zero;

  const int nk = K >> 5;

  // async copy of one 128x32 bf16 tile: 2 x b128 per thread (2 instr/wave)
  auto issueB = [&](int k0, int b) {
    #pragma unroll
    for (int i = 0; i < 2; ++i) {
      int chunk = tid + i * 256;
      int row = chunk >> 2, seg = chunk & 3;
      const __bf16* g = Wb + (size_t)(colBase + row) * K + k0 + seg * 8;
      unsigned l = (unsigned)(size_t)&Bs[b][row][seg * 8];
      asm volatile("global_load_async_to_lds_b128 %0, %1, off"
                   :: "v"(l), "v"(g) : "memory");
    }
  };
  auto issueA16 = [&](int k0, int b) {
    #pragma unroll
    for (int i = 0; i < 2; ++i) {
      int chunk = tid + i * 256;
      int row = chunk >> 2, seg = chunk & 3;
      const __bf16* g = A16 + (size_t)(rowBase + row) * lda + k0 + seg * 8;
      unsigned l = (unsigned)(size_t)&As[b][row][seg * 8];
      asm volatile("global_load_async_to_lds_b128 %0, %1, off"
                   :: "v"(l), "v"(g) : "memory");
    }
  };

  float4 rga[4];
  auto loadA32 = [&](int k0) {
    #pragma unroll
    for (int rr = 0; rr < 4; ++rr)
      rga[rr] = *reinterpret_cast<const float4*>(
          &A32[(size_t)(rowBase + lrow + rr * 32) * lda + k0 + lcol]);
  };
  auto storeA32 = [&](int b) {
    #pragma unroll
    for (int rr = 0; rr < 4; ++rr) {
      Pack4 p;
      p.h[0] = (__bf16)rga[rr].x; p.h[1] = (__bf16)rga[rr].y;
      p.h[2] = (__bf16)rga[rr].z; p.h[3] = (__bf16)rga[rr].w;
      *reinterpret_cast<unsigned long long*>(&As[b][lrow + rr * 32][lcol]) = p.u;
    }
  };

  // prologue: stage tile 0
  issueB(0, 0);
  if (AMODE) issueA16(0, 0); else loadA32(0);

  for (int kt = 0; kt < nk; ++kt) {
    const int buf = kt & 1;
    __syncthreads();                       // WAR: everyone done reading buf^1
    const bool more = (kt + 1 < nk);
    if (more) {
      issueB((kt + 1) * 32, buf ^ 1);
      if (AMODE) issueA16((kt + 1) * 32, buf ^ 1);
    }
    if (!AMODE) {
      storeA32(buf);                       // regs loaded one tile ahead
      if (more) loadA32((kt + 1) * 32);    // prefetch next tile into regs
    }
    // wait for CURRENT tile's asyncs (in-order per wave): leave next tile's
    // (4 for AMODE=1, 2 for AMODE=0) outstanding.
    if (more) {
      if (AMODE) asm volatile("s_wait_asynccnt 0x4" ::: "memory");
      else       asm volatile("s_wait_asynccnt 0x2" ::: "memory");
    } else {
      asm volatile("s_wait_asynccnt 0x0" ::: "memory");
    }
    __syncthreads();                       // cross-wave visibility of buf

    // A fragments (16-bit A 16x32 layout)
    Frag a[2];
    #pragma unroll
    for (int tm = 0; tm < 2; ++tm) {
      int ar = wm * 32 + tm * 16 + l16;
      a[tm].f[0] = *reinterpret_cast<const float4*>(&As[buf][ar][half * 8]);
      a[tm].f[1] = *reinterpret_cast<const float4*>(&As[buf][ar][16 + half * 8]);
    }
    #pragma unroll
    for (int tn = 0; tn < 4; ++tn) {
      int br = wn * 64 + tn * 16 + l16;
      Frag b;
      b.f[0] = *reinterpret_cast<const float4*>(&Bs[buf][br][half * 8]);
      b.f[1] = *reinterpret_cast<const float4*>(&Bs[buf][br][16 + half * 8]);
      #pragma unroll
      for (int tm = 0; tm < 2; ++tm) {
        acc[tm][tn] = __builtin_amdgcn_wmma_f32_16x16x32_bf16(
            false, a[tm].v, false, b.v, (short)0, acc[tm][tn], false, false);
      }
    }
  }

  #pragma unroll
  for (int tm = 0; tm < 2; ++tm)
    #pragma unroll
    for (int tn = 0; tn < 4; ++tn) {
      int col = colBase + wn * 64 + tn * 16 + l16;
      float bv = bias[col];
      #pragma unroll
      for (int j = 0; j < 8; ++j) {
        int row = rowBase + wm * 32 + tm * 16 + half * 8 + j;
        float v = acc[tm][tn][j] + bv;
        if (relu) v = fmaxf(v, 0.f);
        if (Cb) Cb[(size_t)row * ldc + col] = (__bf16)v;
        else    Cf[(size_t)row * ldc + col] = v;
      }
    }
}

// fp32 -> bf16 weight conversion
__global__ __launch_bounds__(256) void k_cvt(const float* __restrict__ s,
                                             __bf16* __restrict__ d, int n)
{
  int i = blockIdx.x * 256 + threadIdx.x;
  if (i < n) d[i] = (__bf16)s[i];
}

// fold out_w @ Wv -> wcomb(bf16), out_w @ bv + out_b -> bcomb(f32)
__global__ __launch_bounds__(256) void k_fuse(
    const float* __restrict__ in_w, const float* __restrict__ in_b,
    const float* __restrict__ out_w, const float* __restrict__ out_b,
    __bf16* __restrict__ wcombbf, float* __restrict__ bcomb)
{
  int n = blockIdx.x, k = threadIdx.x;
  __shared__ float ow[256];
  __shared__ float red[256];
  ow[k] = out_w[n * 256 + k];
  __syncthreads();
  const float* Wv = in_w + 2 * 256 * 256;
  float acc = 0.f;
  for (int j = 0; j < 256; ++j) acc = fmaf(ow[j], Wv[j * 256 + k], acc);
  wcombbf[n * 256 + k] = (__bf16)acc;

  red[k] = ow[k] * in_b[512 + k];
  __syncthreads();
  for (int s = 128; s > 0; s >>= 1) {
    if (k < s) red[k] += red[k + s];
    __syncthreads();
  }
  if (k == 0) bcomb[n] = red[0] + out_b[n];
}

// gate + blend: reads s (f32) and g1 (f32), writes combined as bf16
__global__ __launch_bounds__(256) void k_gate(
    const float* __restrict__ s, const float* __restrict__ g1,
    __bf16* __restrict__ cbf,
    const float* __restrict__ wg2, const float* __restrict__ bg2)
{
  int b = blockIdx.x, t = threadIdx.x;
  __shared__ float red[256];
  __shared__ float sgate[4];
  size_t base = (size_t)b * 1024;
  float w2 = wg2[t];
  #pragma unroll
  for (int m = 0; m < 4; ++m) {
    red[t] = g1[base + m * 256 + t] * w2;
    __syncthreads();
    for (int st = 128; st > 0; st >>= 1) {
      if (t < st) red[t] += red[t + st];
      __syncthreads();
    }
    if (t == 0) sgate[m] = 1.f / (1.f + expf(-(red[0] + bg2[0])));
    __syncthreads();
  }
  float sv[4], cross = 0.f;
  #pragma unroll
  for (int m = 0; m < 4; ++m) { sv[m] = s[base + m * 256 + t]; cross += sv[m]; }
  cross *= 0.25f;
  #pragma unroll
  for (int m = 0; m < 4; ++m) {
    float g = sgate[m];
    cbf[base + m * 256 + t] = (__bf16)(g * sv[m] + (1.f - g) * cross);
  }
}

__global__ __launch_bounds__(256) void k_zero(float* a, float* b) {
  int t = threadIdx.x; a[t] = 0.f; b[t] = 0.f;
}

__global__ __launch_bounds__(256) void k_stats(
    const float* __restrict__ h, float* __restrict__ gsum, float* __restrict__ gsq)
{
  int t = threadIdx.x;
  size_t r0 = (size_t)blockIdx.x * 128;
  float s = 0.f, q = 0.f;
  for (int r = 0; r < 128; ++r) {
    float v = h[(r0 + r) * 256 + t];
    s += v; q = fmaf(v, v, q);
  }
  atomicAdd(&gsum[t], s);
  atomicAdd(&gsq[t], q);
}

__global__ __launch_bounds__(256) void k_finalize(
    const float* gsum, const float* gsq, const float* gamma, const float* beta,
    float* scale, float* shift)
{
  int t = threadIdx.x;
  float mu  = gsum[t] * (1.f / (float)BATCH);
  float var = gsq[t]  * (1.f / (float)BATCH) - mu * mu;
  float r   = rsqrtf(var + 1e-5f);
  float sc  = gamma[t] * r;
  scale[t] = sc;
  shift[t] = beta[t] - mu * sc;
}

__global__ __launch_bounds__(64) void k_out(
    const float* __restrict__ h, const float* __restrict__ scale,
    const float* __restrict__ shift, const float* __restrict__ W2,
    const float* __restrict__ b2, float* __restrict__ out)
{
  int b = blockIdx.x, t = threadIdx.x;
  __shared__ float hn[256];
  for (int i = t; i < 256; i += 64) {
    float v = fmaf(h[(size_t)b * 256 + i], scale[i], shift[i]);
    hn[i] = fmaxf(v, 0.f);
  }
  __syncthreads();
  if (t < 26) {
    float acc = b2[t];
    for (int k = 0; k < 256; ++k) acc = fmaf(hn[k], W2[t * 256 + k], acc);
    out[(size_t)b * 26 + t] = acc;
  }
}

extern "C" void kernel_launch(void* const* d_in, const int* in_sizes, int n_in,
                              void* d_out, int out_size, void* d_ws, size_t ws_size,
                              hipStream_t stream)
{
  const float* xc    = (const float*)d_in[0];
  const float* xb    = (const float*)d_in[1];
  const float* xfc   = (const float*)d_in[2];
  const float* xt    = (const float*)d_in[3];
  const float* Wc    = (const float*)d_in[4];  const float* bc  = (const float*)d_in[5];
  const float* Wb    = (const float*)d_in[6];  const float* bb  = (const float*)d_in[7];
  const float* Wf    = (const float*)d_in[8];  const float* bf  = (const float*)d_in[9];
  const float* Wt    = (const float*)d_in[10]; const float* bt  = (const float*)d_in[11];
  const float* in_w  = (const float*)d_in[12]; const float* in_b = (const float*)d_in[13];
  const float* out_w = (const float*)d_in[14]; const float* out_b = (const float*)d_in[15];
  const float* Wg1   = (const float*)d_in[16]; const float* bg1 = (const float*)d_in[17];
  const float* Wg2   = (const float*)d_in[18]; const float* bg2 = (const float*)d_in[19];
  const float* W1    = (const float*)d_in[20]; const float* b1  = (const float*)d_in[21];
  const float* gamma = (const float*)d_in[22]; const float* beta = (const float*)d_in[23];
  const float* W2    = (const float*)d_in[24]; const float* b2  = (const float*)d_in[25];
  float* out = (float*)d_out;

  // ---- workspace layout ----
  const size_t NFD = (size_t)BATCH * 4 * 256;   // 8,388,608
  float* fp = (float*)d_ws;
  float* sbuf  = fp;             fp += NFD;           // self_out (f32)
  float* g1buf = fp;             fp += NFD;           // gate hidden (f32)
  float* hbuf  = fp;             fp += (size_t)BATCH * 256;
  float* bcomb = fp;             fp += 256;
  float* gsum  = fp;             fp += 256;
  float* gsq   = fp;             fp += 256;
  float* scl   = fp;             fp += 256;
  float* shf   = fp;             fp += 256;
  __bf16* hp = (__bf16*)fp;
  __bf16* fbf     = hp;          hp += NFD;           // f (bf16), reused as combined
  __bf16* wcbf    = hp;          hp += 256 * 2048;
  __bf16* wbbf    = hp;          hp += 256 * 2048;
  __bf16* wfbf    = hp;          hp += 256 * 512;
  __bf16* wtbf    = hp;          hp += 256 * 768;
  __bf16* wg1bf   = hp;          hp += 256 * 256;
  __bf16* w1bf    = hp;          hp += 256 * 1024;
  __bf16* wcombbf = hp;          hp += 256 * 256;
  __bf16* cbf = fbf;                                  // combined overlays f

  dim3 blk(256);
  k_zero<<<1, 256, 0, stream>>>(gsum, gsq);
  k_fuse<<<256, 256, 0, stream>>>(in_w, in_b, out_w, out_b, wcombbf, bcomb);
  k_cvt<<<2048, 256, 0, stream>>>(Wc, wcbf, 256 * 2048);
  k_cvt<<<2048, 256, 0, stream>>>(Wb, wbbf, 256 * 2048);
  k_cvt<<<512,  256, 0, stream>>>(Wf, wfbf, 256 * 512);
  k_cvt<<<768,  256, 0, stream>>>(Wt, wtbf, 256 * 768);
  k_cvt<<<256,  256, 0, stream>>>(Wg1, wg1bf, 256 * 256);
  k_cvt<<<1024, 256, 0, stream>>>(W1, w1bf, 256 * 1024);

  // projections (A fp32, C bf16 -> f[b,m,:])
  gemm_wmma<0><<<dim3(2, 64),  blk, 0, stream>>>(xc, 2048, wcbf, bc, nullptr, fbf + 0,   1024, 2048, 0);
  gemm_wmma<0><<<dim3(2, 64),  blk, 0, stream>>>(xb, 2048, wbbf, bb, nullptr, fbf + 256, 1024, 2048, 0);
  gemm_wmma<0><<<dim3(2, 64),  blk, 0, stream>>>(xfc, 512, wfbf, bf, nullptr, fbf + 512, 1024, 512,  0);
  gemm_wmma<0><<<dim3(2, 64),  blk, 0, stream>>>(xt,  768, wtbf, bt, nullptr, fbf + 768, 1024, 768,  0);
  // self_out = f @ (out_w@Wv).T + folded bias   (A bf16 async, C f32)
  gemm_wmma<1><<<dim3(2, 256), blk, 0, stream>>>(fbf, 256, wcombbf, bcomb, sbuf, nullptr, 256, 256, 0);
  // g1 = relu(self_out @ Wg1.T + bg1)            (A f32, C f32)
  gemm_wmma<0><<<dim3(2, 256), blk, 0, stream>>>(sbuf, 256, wg1bf, bg1, g1buf, nullptr, 256, 256, 1);
  // gate + blend -> combined (bf16, overlays f which is dead)
  k_gate<<<BATCH, 256, 0, stream>>>(sbuf, g1buf, cbf, Wg2, bg2);
  // h = combined.reshape(B,1024) @ W1.T + b1     (A bf16 async, C f32)
  gemm_wmma<1><<<dim3(2, 64),  blk, 0, stream>>>(cbf, 1024, w1bf, b1, hbuf, nullptr, 256, 1024, 0);
  // batchnorm + classifier
  k_stats<<<64, 256, 0, stream>>>(hbuf, gsum, gsq);
  k_finalize<<<1, 256, 0, stream>>>(gsum, gsq, gamma, beta, scl, shf);
  k_out<<<BATCH, 64, 0, stream>>>(hbuf, scl, shf, W2, b2, out);

  (void)in_sizes; (void)n_in; (void)out_size; (void)ws_size;
}
